// SpMiddleResNetFHD_82729660055571
// MI455X (gfx1250) — matrix operand order repaired and checked by reading, not compiled, not run on hardware
//
#include <hip/hip_runtime.h>

typedef __attribute__((ext_vector_type(8)))  _Float16 half8;
typedef __attribute__((ext_vector_type(16))) _Float16 half16;
typedef __attribute__((ext_vector_type(8)))  float    f32x8;

// ---------------------------------------------------------------------------
// Utility kernels
// ---------------------------------------------------------------------------
__global__ void k_fill_h(_Float16* p, int n) {
    int i = blockIdx.x * blockDim.x + threadIdx.x;
    if (i < n) p[i] = (_Float16)0.0f;
}

__global__ void k_fill_f(float* p, int n) {
    int i = blockIdx.x * blockDim.x + threadIdx.x;
    if (i < n) p[i] = 0.0f;
}

// Scatter voxel features into a dense NDHWC f16 grid + occupancy mask.
__global__ void k_scatter(const float* __restrict__ vf, const int* __restrict__ coors,
                          _Float16* __restrict__ act, float* __restrict__ mask,
                          int N, int H, int W, int C) {
    int i = blockIdx.x * blockDim.x + threadIdx.x;
    if (i >= N * C) return;
    int v = i / C, c = i - v * C;
    int z = coors[v * 4 + 1];
    int y = coors[v * 4 + 2];
    int x = coors[v * 4 + 3];
    long base = ((long)z * H + y) * W + x;
    act[base * C + c] = (_Float16)vf[(long)v * C + c];
    if (c == 0) mask[base] = 1.0f;
}

// Fold BN into scale/shift.
__global__ void k_bnprep(const float* __restrict__ g, const float* __restrict__ b,
                         const float* __restrict__ m, const float* __restrict__ v,
                         float* __restrict__ scale, float* __restrict__ shift, int C) {
    int i = blockIdx.x * blockDim.x + threadIdx.x;
    if (i >= C) return;
    float s = g[i] * rsqrtf(v[i] + 1e-3f);
    scale[i] = s;
    shift[i] = b[i] - m[i] * s;
}

// Pre-swizzle OIDHW f32 weights into per-lane B fragments (f16):
// frag layout: [ntileN][kchunk][lane(32)][16 halves], per-lane j -> K = 32*chunk + (j&7) + 8*(lane>=16) + 16*(j>>3)
__global__ void k_wprep(const float* __restrict__ Wt, _Float16* __restrict__ frag,
                        int Cin, int Cout, int taps, int nchunk) {
    int ntN = Cout >> 4;
    long total = (long)ntN * nchunk * 512;
    long e = (long)blockIdx.x * blockDim.x + threadIdx.x;
    if (e >= total) return;
    int j    = (int)(e & 15);
    int lane = (int)((e >> 4) & 31);
    long t   = e >> 9;
    int c    = (int)(t % nchunk);
    int n    = (int)(t / nchunk);
    int K    = c * 32 + (j & 7) + ((lane >> 4) & 1) * 8 + (j >> 3) * 16;
    int co   = (lane & 15) + 16 * n;
    float val = 0.0f;
    if (K < taps * Cin) {
        int tap = K / Cin;
        int ci  = K - tap * Cin;
        val = Wt[((long)co * Cin + ci) * taps + tap];  // OIDHW: tap is the inner DHW index
    }
    frag[e] = (_Float16)val;
}

// reduce_window(max, init=0) for the occupancy mask.
__global__ void k_maskpool(const float* __restrict__ in, float* __restrict__ out,
                           int Din, int Hin, int Win, int Dout, int Hout, int Wout,
                           int kd, int kh, int kw, int sz, int sy, int sx,
                           int pz, int py, int px) {
    int m = blockIdx.x * blockDim.x + threadIdx.x;
    int V = Dout * Hout * Wout;
    if (m >= V) return;
    int z = m / (Hout * Wout);
    int r = m - z * (Hout * Wout);
    int y = r / Wout;
    int x = r - y * Wout;
    float mx = 0.0f;
    for (int a = 0; a < kd; ++a) {
        int zi = z * sz - pz + a;
        if ((unsigned)zi >= (unsigned)Din) continue;
        for (int b = 0; b < kh; ++b) {
            int yi = y * sy - py + b;
            if ((unsigned)yi >= (unsigned)Hin) continue;
            for (int cc = 0; cc < kw; ++cc) {
                int xi = x * sx - px + cc;
                if ((unsigned)xi >= (unsigned)Win) continue;
                float vv = in[((long)zi * Hin + yi) * Win + xi];
                mx = fmaxf(mx, vv);
            }
        }
    }
    out[m] = mx;
}

// ---------------------------------------------------------------------------
// Implicit-GEMM 3D conv with WMMA f16->f32, fused BN + (residual) + ReLU + mask.
// Templated on channel counts + kernel dims so the K-chunk loop fully unrolls:
// tap index / kernel offsets / channel bases are compile-time constants; the
// per-tap address offsets are SGPR expressions; only a base offset and a tap
// validity bitmask are per-lane.  One wave = 16 output voxels x COUT channels.
// ---------------------------------------------------------------------------
template <int CIN, int COUT, int KD, int KH, int KW>
__global__ __launch_bounds__(128)
void k_conv_wmma(const _Float16* __restrict__ in, _Float16* __restrict__ out,
                 const _Float16* __restrict__ wfrag,
                 const float* __restrict__ scale, const float* __restrict__ shift,
                 const float* __restrict__ mask, const _Float16* __restrict__ res,
                 int Din, int Hin, int Win,
                 int Dout, int Hout, int Wout,
                 int sz, int sy, int sx, int pz, int py, int px) {
    constexpr int TAPS = KD * KH * KW;
    constexpr int KTOT = TAPS * CIN;
    constexpr int NCH  = (KTOT + 31) / 32;
    constexpr int NTN  = COUT / 16;

    const int lane   = threadIdx.x & 31;
    const int halfid = lane >> 4;
    const int nlane  = lane & 15;
    const int Vout   = Dout * Hout * Wout;
    const int tile   = blockIdx.x * (blockDim.x >> 5) + (threadIdx.x >> 5);
    if (tile * 16 >= Vout) return;               // wave-uniform exit -> EXEC stays full

    // Per-lane output voxel for the A-matrix row (M = lane&15)
    const int mA  = tile * 16 + nlane;
    const bool mAv = (mA < Vout);
    int t0 = mAv ? mA : 0;
    const int zo = t0 / (Hout * Wout);
    int rr = t0 - zo * (Hout * Wout);
    const int yo = rr / Wout;
    const int xo = rr - yo * Wout;
    const int zb = zo * sz - pz;
    const int yb = yo * sy - py;
    const int xb = xo * sx - px;

    // Scalar (uniform) strides in elements; per-lane base offset + validity bits.
    const int sWl  = Win * CIN;
    const int sHWl = Hin * sWl;
    const int b0   = (zb * Hin + yb) * sWl + xb * CIN;
    unsigned validMask = 0;
    if (mAv) {
#pragma unroll
        for (int t = 0; t < TAPS; ++t) {
            const int kz = t / (KH * KW);
            const int ky = (t / KW) % KH;
            const int kx = t % KW;
            const bool ok = (unsigned)(zb + kz) < (unsigned)Din &&
                            (unsigned)(yb + ky) < (unsigned)Hin &&
                            (unsigned)(xb + kx) < (unsigned)Win;
            validMask |= ok ? (1u << t) : 0u;
        }
    }

    const int ciH = halfid * 8;                  // lane-half channel offset
    f32x8 acc[NTN] = {};

#pragma unroll
    for (int c = 0; c < NCH; ++c) {
        half8 g[2];
#pragma unroll
        for (int gi = 0; gi < 2; ++gi) {
            constexpr int dummy = 0; (void)dummy;
            const int kbase = c * 32 + gi * 16;          // compile-time
            const int tap   = kbase / CIN;               // compile-time
            const int ci0   = kbase % CIN;               // compile-time
            half8 v = {};
            if (tap < TAPS) {                            // compile-time prune
                const int kz = tap / (KH * KW);
                const int ky = (tap / KW) % KH;
                const int kx = tap % KW;
                if ((validMask >> tap) & 1u) {
                    v = *reinterpret_cast<const half8*>(
                        in + b0 + kz * sHWl + ky * sWl + kx * CIN + ci0 + ciH);
                }
            }
            g[gi] = v;
        }
        half16 A;
#pragma unroll
        for (int i = 0; i < 8; ++i) { A[i] = g[0][i]; A[8 + i] = g[1][i]; }

        const _Float16* wc = wfrag + (c * 32 + lane) * 16;
        __builtin_prefetch(wc + 512, 0, 0);      // next chunk's B fragments
#pragma unroll
        for (int n = 0; n < NTN; ++n) {
            half16 B = *reinterpret_cast<const half16*>(wc + n * (NCH * 512));
            acc[n] = __builtin_amdgcn_wmma_f32_16x16x32_f16(
                false, A, false, B, (short)0, acc[n], false, false);
        }
    }

    // Epilogue: BN scale/shift (+residual) -> ReLU -> mask -> f16 store.
    float scl[NTN], shf[NTN];
#pragma unroll
    for (int n = 0; n < NTN; ++n) {
        const int co = nlane + 16 * n;
        scl[n] = scale[co];
        shf[n] = shift[co];
    }
#pragma unroll
    for (int r = 0; r < 8; ++r) {
        const int m = tile * 16 + halfid * 8 + r;   // C/D: M = r + 8*(lane>=16)
        if (m < Vout) {
            const float mk = mask[m];
#pragma unroll
            for (int n = 0; n < NTN; ++n) {
                const int co = nlane + 16 * n;
                float vv = acc[n][r] * scl[n] + shf[n];
                if (res) vv += (float)res[(long)m * COUT + co];
                vv = fmaxf(vv, 0.0f) * mk;
                out[(long)m * COUT + co] = (_Float16)vv;
            }
        }
    }
}

// (1,C,Dd,H,W) f16 NDHWC -> (1, C*Dd, H, W) f32 NCHW
__global__ void k_finalize(const _Float16* __restrict__ act, float* __restrict__ out,
                           int Dd, int Hh, int Ww, int C) {
    int i = blockIdx.x * blockDim.x + threadIdx.x;
    int total = C * Dd * Hh * Ww;
    if (i >= total) return;
    int x = i % Ww;
    int t = i / Ww;
    int y = t % Hh; t /= Hh;
    int z = t % Dd;
    int c = t / Dd;
    out[i] = (float)act[(((long)z * Hh + y) * Ww + x) * C + c];
}

// ---------------------------------------------------------------------------
// Host orchestration
// ---------------------------------------------------------------------------
extern "C" void kernel_launch(void* const* d_in, const int* in_sizes, int n_in,
                              void* d_out, int out_size, void* d_ws, size_t ws_size,
                              hipStream_t stream) {
    (void)in_sizes; (void)n_in; (void)out_size; (void)ws_size;

    const float* vf    = (const float*)d_in[0];
    const int*   coors = (const int*)d_in[1];
    const int    NVOX  = 60000;

    // Grid pyramid
    const int D0 = 41, H0 = 200, W0 = 176;
    const int D1 = 21, H1 = 100, W1 = 88;
    const int D2 = 11, H2 = 50,  W2 = 44;
    const int D3 = 6,  H3 = 25,  W3 = 22;
    const int D4 = 2,  H4 = 25,  W4 = 22;
    const int V0 = D0 * H0 * W0, V1 = D1 * H1 * W1, V2 = D2 * H2 * W2;
    const int V3 = D3 * H3 * W3, V4 = D4 * H4 * W4;

    // bump allocator on workspace
    char* base = (char*)d_ws;
    size_t off = 0;
    auto alloc = [&](size_t bytes) -> char* {
        char* p = base + off;
        off += (bytes + 255) & ~(size_t)255;
        return p;
    };

    _Float16* actA = (_Float16*)alloc((size_t)V0 * 16 * sizeof(_Float16));
    _Float16* actB = (_Float16*)alloc((size_t)V0 * 16 * sizeof(_Float16));
    float* m0 = (float*)alloc((size_t)V0 * 4);
    float* m1 = (float*)alloc((size_t)V1 * 4);
    float* m2 = (float*)alloc((size_t)V2 * 4);
    float* m3 = (float*)alloc((size_t)V3 * 4);
    float* m4 = (float*)alloc((size_t)V4 * 4);

    // Layer table: {weight input idx (bn g/b/m/v follow at +1..+4), Cin, Cout, kd,kh,kw}
    struct LInfo { int widx, cin, cout, kd, kh, kw; };
    const LInfo L[17] = {
        { 3, 16, 16, 3, 3, 3},  // 0  stem
        { 8, 16, 16, 3, 3, 3},  // 1  blkA.conv1
        {13, 16, 16, 3, 3, 3},  // 2  blkA.conv2
        {18, 16, 32, 3, 3, 3},  // 3  down1
        {23, 32, 32, 3, 3, 3},  // 4  blkB.conv1
        {28, 32, 32, 3, 3, 3},  // 5  blkB.conv2
        {33, 32, 64, 3, 3, 3},  // 6  down2
        {38, 64, 64, 3, 3, 3},  // 7  blkC1.conv1
        {43, 64, 64, 3, 3, 3},  // 8  blkC1.conv2
        {48, 64, 64, 3, 3, 3},  // 9  blkC2.conv1
        {53, 64, 64, 3, 3, 3},  // 10 blkC2.conv2
        {58, 64, 64, 3, 3, 3},  // 11 down3
        {63, 64, 64, 3, 3, 3},  // 12 blkD1.conv1
        {68, 64, 64, 3, 3, 3},  // 13 blkD1.conv2
        {73, 64, 64, 3, 3, 3},  // 14 blkD2.conv1
        {78, 64, 64, 3, 3, 3},  // 15 blkD2.conv2
        {83, 64, 64, 3, 1, 1},  // 16 down4
    };

    _Float16* frag[17];
    float* bns[17];
    float* bnh[17];
    for (int l = 0; l < 17; ++l) {
        int taps = L[l].kd * L[l].kh * L[l].kw;
        int nch  = (taps * L[l].cin + 31) / 32;
        int ntn  = L[l].cout / 16;
        size_t nel = (size_t)ntn * nch * 512;
        frag[l] = (_Float16*)alloc(nel * sizeof(_Float16));
        bns[l]  = (float*)alloc((size_t)L[l].cout * 4);
        bnh[l]  = (float*)alloc((size_t)L[l].cout * 4);
        k_wprep<<<(int)((nel + 255) / 256), 256, 0, stream>>>(
            (const float*)d_in[L[l].widx], frag[l], L[l].cin, L[l].cout, taps, nch);
        k_bnprep<<<1, 64, 0, stream>>>(
            (const float*)d_in[L[l].widx + 1], (const float*)d_in[L[l].widx + 2],
            (const float*)d_in[L[l].widx + 3], (const float*)d_in[L[l].widx + 4],
            bns[l], bnh[l], L[l].cout);
    }

    // Dense grid + mask init, voxel scatter
    {
        int n = V0 * 16;
        k_fill_h<<<(n + 255) / 256, 256, 0, stream>>>(actA, n);
        k_fill_f<<<(V0 + 255) / 256, 256, 0, stream>>>(m0, V0);
        k_scatter<<<(NVOX * 16 + 255) / 256, 256, 0, stream>>>(vf, coors, actA, m0,
                                                               NVOX, H0, W0, 16);
    }

    // Mask pyramid (max pool matching each down-conv)
    k_maskpool<<<(V1 + 255) / 256, 256, 0, stream>>>(m0, m1, D0, H0, W0, D1, H1, W1,
                                                     3, 3, 3, 2, 2, 2, 1, 1, 1);
    k_maskpool<<<(V2 + 255) / 256, 256, 0, stream>>>(m1, m2, D1, H1, W1, D2, H2, W2,
                                                     3, 3, 3, 2, 2, 2, 1, 1, 1);
    k_maskpool<<<(V3 + 255) / 256, 256, 0, stream>>>(m2, m3, D2, H2, W2, D3, H3, W3,
                                                     3, 3, 3, 2, 2, 2, 1, 1, 1);
    k_maskpool<<<(V4 + 255) / 256, 256, 0, stream>>>(m3, m4, D3, H3, W3, D4, H4, W4,
                                                     3, 1, 1, 2, 1, 1, 0, 0, 0);

    auto conv = [&](const _Float16* in, _Float16* out, int l,
                    const float* mk, const _Float16* res,
                    int Din, int Hin, int Win, int Dout, int Hout, int Wout,
                    int sz, int sy, int sx, int pz, int py, int px) {
        int Vout = Dout * Hout * Wout;
        int tiles = (Vout + 15) / 16;
        int blocks = (tiles + 3) / 4;
        dim3 gr(blocks), bl(128);
        switch (l) {
        case 0: case 1: case 2:
            k_conv_wmma<16, 16, 3, 3, 3><<<gr, bl, 0, stream>>>(
                in, out, frag[l], bns[l], bnh[l], mk, res,
                Din, Hin, Win, Dout, Hout, Wout, sz, sy, sx, pz, py, px);
            break;
        case 3:
            k_conv_wmma<16, 32, 3, 3, 3><<<gr, bl, 0, stream>>>(
                in, out, frag[l], bns[l], bnh[l], mk, res,
                Din, Hin, Win, Dout, Hout, Wout, sz, sy, sx, pz, py, px);
            break;
        case 4: case 5:
            k_conv_wmma<32, 32, 3, 3, 3><<<gr, bl, 0, stream>>>(
                in, out, frag[l], bns[l], bnh[l], mk, res,
                Din, Hin, Win, Dout, Hout, Wout, sz, sy, sx, pz, py, px);
            break;
        case 6:
            k_conv_wmma<32, 64, 3, 3, 3><<<gr, bl, 0, stream>>>(
                in, out, frag[l], bns[l], bnh[l], mk, res,
                Din, Hin, Win, Dout, Hout, Wout, sz, sy, sx, pz, py, px);
            break;
        case 16:
            k_conv_wmma<64, 64, 3, 1, 1><<<gr, bl, 0, stream>>>(
                in, out, frag[l], bns[l], bnh[l], mk, res,
                Din, Hin, Win, Dout, Hout, Wout, sz, sy, sx, pz, py, px);
            break;
        default:
            k_conv_wmma<64, 64, 3, 3, 3><<<gr, bl, 0, stream>>>(
                in, out, frag[l], bns[l], bnh[l], mk, res,
                Din, Hin, Win, Dout, Hout, Wout, sz, sy, sx, pz, py, px);
            break;
        }
    };

    // stem: A -> B            (x now in B)
    conv(actA, actB, 0,  m0, nullptr, D0, H0, W0, D0, H0, W0, 1, 1, 1, 1, 1, 1);
    // blkA: conv1 B->A; conv2 A->B (+res B)
    conv(actB, actA, 1,  m0, nullptr, D0, H0, W0, D0, H0, W0, 1, 1, 1, 1, 1, 1);
    conv(actA, actB, 2,  m0, actB,    D0, H0, W0, D0, H0, W0, 1, 1, 1, 1, 1, 1);
    // down1: B(L0,16) -> A(L1,32)
    conv(actB, actA, 3,  m1, nullptr, D0, H0, W0, D1, H1, W1, 2, 2, 2, 1, 1, 1);
    // blkB: conv1 A->B; conv2 B->A (+res A)
    conv(actA, actB, 4,  m1, nullptr, D1, H1, W1, D1, H1, W1, 1, 1, 1, 1, 1, 1);
    conv(actB, actA, 5,  m1, actA,    D1, H1, W1, D1, H1, W1, 1, 1, 1, 1, 1, 1);
    // down2: A(L1,32) -> B(L2,64)
    conv(actA, actB, 6,  m2, nullptr, D1, H1, W1, D2, H2, W2, 2, 2, 2, 1, 1, 1);
    // blkC1: conv1 B->A; conv2 A->B (+res B)
    conv(actB, actA, 7,  m2, nullptr, D2, H2, W2, D2, H2, W2, 1, 1, 1, 1, 1, 1);
    conv(actA, actB, 8,  m2, actB,    D2, H2, W2, D2, H2, W2, 1, 1, 1, 1, 1, 1);
    // blkC2: conv1 B->A; conv2 A->B (+res B)
    conv(actB, actA, 9,  m2, nullptr, D2, H2, W2, D2, H2, W2, 1, 1, 1, 1, 1, 1);
    conv(actA, actB, 10, m2, actB,    D2, H2, W2, D2, H2, W2, 1, 1, 1, 1, 1, 1);
    // down3: B(L2) -> A(L3)
    conv(actB, actA, 11, m3, nullptr, D2, H2, W2, D3, H3, W3, 2, 2, 2, 1, 1, 1);
    // blkD1: conv1 A->B; conv2 B->A (+res A)
    conv(actA, actB, 12, m3, nullptr, D3, H3, W3, D3, H3, W3, 1, 1, 1, 1, 1, 1);
    conv(actB, actA, 13, m3, actA,    D3, H3, W3, D3, H3, W3, 1, 1, 1, 1, 1, 1);
    // blkD2: conv1 A->B; conv2 B->A (+res A)
    conv(actA, actB, 14, m3, nullptr, D3, H3, W3, D3, H3, W3, 1, 1, 1, 1, 1, 1);
    conv(actB, actA, 15, m3, actA,    D3, H3, W3, D3, H3, W3, 1, 1, 1, 1, 1, 1);
    // down4: A(L3) -> B(L4), kernel (3,1,1) stride (2,1,1) valid
    conv(actA, actB, 16, m4, nullptr, D3, H3, W3, D4, H4, W4, 2, 1, 1, 0, 0, 0);

    // (64,2,25,22) NDHWC f16 -> (128,25,22) NCHW f32
    int total = 64 * D4 * H4 * W4;
    k_finalize<<<(total + 255) / 256, 256, 0, stream>>>(actB, (float*)d_out,
                                                        D4, H4, W4, 64);
}
